// Onet_34548716929431
// MI455X (gfx1250) — compile-verified, matrix operand order checked by reading, not run on hardware
//
#include <hip/hip_runtime.h>
#include <hip/hip_bf16.h>
#include <stdint.h>

typedef __attribute__((ext_vector_type(16))) _Float16 v16h;
typedef __attribute__((ext_vector_type(8)))  float    v8f;

typedef union { v16h h; uint32_t u[8]; } frag16;

#define NBOX 2048
#define IMW  1024
#define IMH  1024

__device__ __forceinline__ uint32_t ordf(float f) {
    uint32_t u = __float_as_uint(f);
    return (u & 0x80000000u) ? ~u : (u | 0x80000000u);
}
__device__ __forceinline__ float unordf(uint32_t u) {
    return (u & 0x80000000u) ? __uint_as_float(u & 0x7FFFFFFFu) : __uint_as_float(~u);
}

// ---------------------------------------------------------------------------
// K0: weight prep.  dst[n*K + k] = (f16) src[k*N + n]   (HWIO -> [n][k])
// ---------------------------------------------------------------------------
__global__ void wprep_transpose(const float* __restrict__ src, _Float16* __restrict__ dst,
                                int K, int N) {
    int idx = blockIdx.x * blockDim.x + threadIdx.x;
    if (idx < K * N) {
        int n = idx / K, k = idx % K;
        dst[idx] = (_Float16)src[k * N + n];
    }
}
__global__ void wprep_direct(const float* __restrict__ src, _Float16* __restrict__ dst, int total) {
    int idx = blockIdx.x * blockDim.x + threadIdx.x;
    if (idx < total) dst[idx] = (_Float16)src[idx];
}

// ---------------------------------------------------------------------------
// K1: per-crop CNN.  One 256-thread block (8 waves) per bbox.
//   crop(48x48x3) -> conv1+prelu+pool -> pool1(23x23x32)
//   conv2 WMMA -> pool2(10x10x64) -> conv3 WMMA -> pool3(4x4x64)
//   conv4 WMMA -> h4g[crop][c*9+pos]  (NCHW flatten, f16, global ws)
// ---------------------------------------------------------------------------
// LDS layout (overlaid regions), total 60096 B:
//   R1 @0      (33856): pool1  -> pool2h -> pool3h
//   R2 @33856  (25600): crop(13824)+w1(3456) -> pool2u -> pool3u
//   tables @59456: xs[48], ys[48], b1[32], a1[32]
__global__ __launch_bounds__(256) void onet_cnn(
    const float* __restrict__ bboxes, const float* __restrict__ image,
    const float* __restrict__ c1w, const float* __restrict__ c1b, const float* __restrict__ p1,
    const float* __restrict__ c2b, const float* __restrict__ p2,
    const float* __restrict__ c3b, const float* __restrict__ p3,
    const float* __restrict__ c4b, const float* __restrict__ p4,
    const _Float16* __restrict__ Bw2, const _Float16* __restrict__ Bw3,
    const _Float16* __restrict__ Bw4, _Float16* __restrict__ h4g)
{
    __shared__ __align__(16) unsigned char smem[60096];
    _Float16* pool1  = (_Float16*)(smem);           // 23*23*32
    _Float16* pool2h = (_Float16*)(smem);           // 10*10*64 (reuse R1)
    _Float16* pool3h = (_Float16*)(smem);           // 4*4*64   (reuse R1)
    _Float16* crop   = (_Float16*)(smem + 33856);   // 48*48*3
    float*    w1     = (float*)   (smem + 47680);   // 864
    uint32_t* pool2u = (uint32_t*)(smem + 33856);   // 6400 (reuse R2)
    uint32_t* pool3u = (uint32_t*)(smem + 33856);   // 1024 (reuse R2)
    int*   sxs = (int*)(smem + 59456);
    int*   sys = (int*)(smem + 59648);
    float* b1s = (float*)(smem + 59840);
    float* a1s = (float*)(smem + 59968);

    const int tid  = threadIdx.x;
    const int lane = tid & 31;
    const int wv   = tid >> 5;
    const int box  = blockIdx.x;

    // ---- step a: sampling tables + conv1 weights ----
    {
        float b0 = bboxes[box*4+0], bb1 = bboxes[box*4+1];
        float b2 = bboxes[box*4+2], b3  = bboxes[box*4+3];
        float bx1 = fmaxf(b0, 0.f), by1 = fmaxf(bb1, 0.f);
        float bx2 = fminf(b2, (float)IMW), by2 = fminf(b3, (float)IMH);
        int i0 = (int)bx1, i1 = (int)by1, i2 = (int)bx2, i3 = (int)by2;
        int cw = max(i2 - i0, 1), ch = max(i3 - i1, 1);
        if (tid < 48) { sxs[tid] = i0 + (tid*cw)/48; sys[tid] = i1 + (tid*ch)/48; }
        for (int i = tid; i < 864; i += 256) w1[i] = c1w[i];
        if (tid < 32) { b1s[tid] = c1b[tid]; a1s[tid] = p1[tid]; }
    }
    __syncthreads();

    // ---- step b: gather crop, normalize, f16 ----
    for (int i = tid; i < 48*48*3; i += 256) {
        int c = i % 3, x = (i/3) % 48, y = i / 144;
        float v = (image[(sys[y]*IMW + sxs[x])*3 + c] - 127.5f) * 0.0078125f;
        crop[i] = (_Float16)v;
    }
    __syncthreads();

    // ---- step c: conv1(3x3,3->32)+bias, pool(3,2,pad1), prelu ----
    for (int idx = tid; idx < 23*23*32; idx += 256) {
        int oc = idx & 31, px = (idx >> 5) % 23, py = idx / 736;
        float m = -__builtin_inff();
        for (int wy = 0; wy < 3; ++wy) {
            int cy = 2*py - 1 + wy; if (cy < 0 || cy > 45) continue;
            for (int wx = 0; wx < 3; ++wx) {
                int cx = 2*px - 1 + wx; if (cx < 0 || cx > 45) continue;
                float acc = b1s[oc];
                for (int ky = 0; ky < 3; ++ky)
                    for (int kx = 0; kx < 3; ++kx) {
                        const _Float16* cp = &crop[((cy+ky)*48 + (cx+kx))*3];
                        const float* wp = &w1[((ky*3+kx)*3)*32 + oc];
                        acc += (float)cp[0]*wp[0] + (float)cp[1]*wp[32] + (float)cp[2]*wp[64];
                    }
                m = fmaxf(m, acc);
            }
        }
        float v = (m >= 0.f) ? m : a1s[oc]*m;
        pool1[(py*23+px)*32 + oc] = (_Float16)v;
    }
    __syncthreads();
    for (int i = tid; i < 6400; i += 256) pool2u[i] = 0u;   // ordered -inf sentinel
    __syncthreads();

    const int hi8  = (lane >> 4) << 3;
    const int l15  = lane & 15;

    // ---- step d: conv2(3x3,32->64) via WMMA, scatter-max pool(3,2) ----
    for (int pair = wv; pair < 112; pair += 8) {
        int mt = pair >> 2, nt = pair & 3;
        int mrow = mt*16 + l15; if (mrow > 440) mrow = 440;
        int oy = mrow / 21, ox = mrow % 21;
        int ncol = nt*16 + l15;
        const uint32_t* Bp = (const uint32_t*)(Bw2 + ncol*288);
        v8f acc = {};
        for (int s = 0; s < 9; ++s) {
            int ky = s/3, kx = s%3;
            const uint32_t* Ap = (const uint32_t*)(pool1 + ((oy+ky)*23 + (ox+kx))*32);
            frag16 a, b;
            #pragma unroll
            for (int r = 0; r < 8; ++r) {
                int kk = ((r & 3) << 1) + ((r >> 2) << 4) + hi8;
                a.u[r] = Ap[kk >> 1];
                b.u[r] = Bp[(s*32 + kk) >> 1];
            }
            acc = __builtin_amdgcn_wmma_f32_16x16x32_f16(false, a.h, false, b.h,
                                                         (short)0, acc, false, false);
        }
        int n = nt*16 + l15;
        #pragma unroll
        for (int j = 0; j < 8; ++j) {
            int m = mt*16 + j + hi8;
            if (m < 441) {
                int oy2 = m/21, ox2 = m%21;
                uint32_t uv = ordf(acc[j]);
                int pylo = (oy2 >= 2) ? ((oy2-1)>>1) : 0, pyhi = min(9, oy2>>1);
                int pxlo = (ox2 >= 2) ? ((ox2-1)>>1) : 0, pxhi = min(9, ox2>>1);
                for (int py = pylo; py <= pyhi; ++py)
                    for (int px = pxlo; px <= pxhi; ++px)
                        atomicMax(&pool2u[(py*10+px)*64 + n], uv);
            }
        }
    }
    __syncthreads();
    // convert pool2 (bias + prelu commute with max)
    for (int i = tid; i < 6400; i += 256) {
        int n = i & 63;
        float f = unordf(pool2u[i]) + c2b[n];
        pool2h[i] = (_Float16)((f >= 0.f) ? f : p2[n]*f);
    }
    __syncthreads();
    for (int i = tid; i < 1024; i += 256) pool3u[i] = 0u;
    __syncthreads();

    // ---- step f: conv3(3x3,64->64) via WMMA, scatter-max pool(2,2) ----
    for (int pair = wv; pair < 16; pair += 8) {
        int mt = pair >> 2, nt = pair & 3;
        int m = mt*16 + l15;
        int oy = m >> 3, ox = m & 7;
        int ncol = nt*16 + l15;
        const uint32_t* Bp = (const uint32_t*)(Bw3 + ncol*576);
        v8f acc = {};
        for (int s = 0; s < 18; ++s) {
            int kk9 = s >> 1, ky = kk9/3, kx = kk9%3, icb = (s & 1) << 5;
            const uint32_t* Ap = (const uint32_t*)(pool2h + ((oy+ky)*10 + (ox+kx))*64 + icb);
            frag16 a, b;
            #pragma unroll
            for (int r = 0; r < 8; ++r) {
                int kk = ((r & 3) << 1) + ((r >> 2) << 4) + hi8;
                a.u[r] = Ap[kk >> 1];
                b.u[r] = Bp[(s*32 + kk) >> 1];
            }
            acc = __builtin_amdgcn_wmma_f32_16x16x32_f16(false, a.h, false, b.h,
                                                         (short)0, acc, false, false);
        }
        int n = nt*16 + l15;
        #pragma unroll
        for (int j = 0; j < 8; ++j) {
            int m2 = mt*16 + j + hi8;
            int oy2 = m2 >> 3, ox2 = m2 & 7;
            atomicMax(&pool3u[((oy2>>1)*4 + (ox2>>1))*64 + n], ordf(acc[j]));
        }
    }
    __syncthreads();
    for (int i = tid; i < 1024; i += 256) {
        int n = i & 63;
        float f = unordf(pool3u[i]) + c3b[n];
        pool3h[i] = (_Float16)((f >= 0.f) ? f : p3[n]*f);
    }
    __syncthreads();

    // ---- step g: conv4(2x2,64->128) via WMMA -> h4g (NCHW flatten) ----
    {
        int nt = wv;                       // 8 waves == 8 N-tiles
        int m = l15; int mm = (m < 9) ? m : 0;
        int oy = mm / 3, ox = mm % 3;
        int ncol = nt*16 + l15;
        const uint32_t* Bp = (const uint32_t*)(Bw4 + ncol*256);
        v8f acc = {};
        for (int s = 0; s < 8; ++s) {
            int kk4 = s >> 1, ky = kk4 >> 1, kx = kk4 & 1, icb = (s & 1) << 5;
            const uint32_t* Ap = (const uint32_t*)(pool3h + ((oy+ky)*4 + (ox+kx))*64 + icb);
            frag16 a, b;
            #pragma unroll
            for (int r = 0; r < 8; ++r) {
                int kk = ((r & 3) << 1) + ((r >> 2) << 4) + hi8;
                a.u[r] = Ap[kk >> 1];
                b.u[r] = Bp[(s*32 + kk) >> 1];
            }
            acc = __builtin_amdgcn_wmma_f32_16x16x32_f16(false, a.h, false, b.h,
                                                         (short)0, acc, false, false);
        }
        int n = nt*16 + l15;
        #pragma unroll
        for (int j = 0; j < 8; ++j) {
            int m2 = j + hi8;
            if (m2 < 9) {
                float f = acc[j] + c4b[n];
                f = (f >= 0.f) ? f : p4[n]*f;
                h4g[(size_t)box*1152 + n*9 + m2] = (_Float16)f;
            }
        }
    }
}

// ---------------------------------------------------------------------------
// K2: fc5 (2048x256x1152 GEMM, WMMA) + prelu + fc6a softmax + fc6b regression
// block = 256 threads handles 16 crops; grid = 128
// ---------------------------------------------------------------------------
__global__ __launch_bounds__(256) void onet_fc(
    const _Float16* __restrict__ h4g, const _Float16* __restrict__ Bw5,
    const float* __restrict__ fc5b, const float* __restrict__ p5,
    const float* __restrict__ f6aw, const float* __restrict__ f6ab,
    const float* __restrict__ f6bw, const float* __restrict__ f6bb,
    const float* __restrict__ bboxes,
    float* __restrict__ score_g, float* __restrict__ obox_g)
{
    __shared__ float f5[16*256];
    __shared__ float o6[16*6];
    const int tid = threadIdx.x, lane = tid & 31, wv = tid >> 5;
    const int hi8 = (lane >> 4) << 3, l15 = lane & 15;
    const int mrow = blockIdx.x*16 + l15;
    const uint32_t* Ap = (const uint32_t*)(h4g + (size_t)mrow*1152);

    for (int t = 0; t < 2; ++t) {
        int nt = wv + t*8;
        int ncol = nt*16 + l15;
        const uint32_t* Bp = (const uint32_t*)(Bw5 + (size_t)ncol*1152);
        v8f acc = {};
        for (int s = 0; s < 36; ++s) {
            frag16 a, b;
            #pragma unroll
            for (int r = 0; r < 8; ++r) {
                int kk = ((r & 3) << 1) + ((r >> 2) << 4) + hi8;
                int k = s*32 + kk;
                a.u[r] = Ap[k >> 1];
                b.u[r] = Bp[k >> 1];
            }
            acc = __builtin_amdgcn_wmma_f32_16x16x32_f16(false, a.h, false, b.h,
                                                         (short)0, acc, false, false);
        }
        int n = nt*16 + l15;
        #pragma unroll
        for (int j = 0; j < 8; ++j) {
            int m = j + hi8;
            float f = acc[j] + fc5b[n];
            f5[m*256 + n] = (f >= 0.f) ? f : p5[n]*f;
        }
    }
    __syncthreads();
    if (tid < 96) {
        int cl = tid / 6, o = tid % 6;
        const float* wp = (o < 2) ? (f6aw + o*256) : (f6bw + (o-2)*256);
        float s = (o < 2) ? f6ab[o] : f6bb[o-2];
        const float* hp = &f5[cl*256];
        for (int k = 0; k < 256; ++k) s += hp[k]*wp[k];
        o6[cl*6 + o] = s;
    }
    __syncthreads();
    if (tid < 16) {
        int crop = blockIdx.x*16 + tid;
        float c0 = o6[tid*6+0], c1 = o6[tid*6+1];
        float score = 1.f / (1.f + expf(c0 - c1));     // softmax[:,1]
        score_g[crop] = score;
        float b0 = bboxes[crop*4+0], b1 = bboxes[crop*4+1];
        float b2 = bboxes[crop*4+2], b3 = bboxes[crop*4+3];
        float bx1 = fmaxf(b0, 0.f), by1 = fmaxf(b1, 0.f);
        float bx2 = fminf(b2, (float)IMW), by2 = fminf(b3, (float)IMH);
        float w = bx2 - bx1, h = by2 - by1;
        obox_g[crop*4+0] = bx1 + o6[tid*6+2]*w;
        obox_g[crop*4+1] = by1 + o6[tid*6+3]*h;
        obox_g[crop*4+2] = bx2 + o6[tid*6+4]*w;
        obox_g[crop*4+3] = by2 + o6[tid*6+5]*h;
    }
}

// ---------------------------------------------------------------------------
// K3: bitonic sort by (score desc, index asc); order_g[i] = orig index
// ---------------------------------------------------------------------------
__global__ __launch_bounds__(1024) void nms_sort(const float* __restrict__ score_g,
                                                 uint32_t* __restrict__ order_g)
{
    __shared__ unsigned long long key[NBOX];
    const int tid = threadIdx.x;
    for (int i = tid; i < NBOX; i += 1024) {
        float s = score_g[i];
        float sm = (s >= 0.5f) ? s : -__builtin_inff();
        key[i] = ((unsigned long long)(~ordf(sm)) << 32) | (uint32_t)i;
    }
    __syncthreads();
    for (int k = 2; k <= NBOX; k <<= 1) {
        for (int j = k >> 1; j > 0; j >>= 1) {
            for (int i = tid; i < NBOX; i += 1024) {
                int p = i ^ j;
                if (p > i) {
                    bool asc = ((i & k) == 0);
                    unsigned long long a = key[i], b = key[p];
                    if ((a > b) == asc) { key[i] = b; key[p] = a; }
                }
            }
            __syncthreads();
        }
    }
    for (int i = tid; i < NBOX; i += 1024) order_g[i] = (uint32_t)key[i];
}

// ---------------------------------------------------------------------------
// K4: suppression mask.  mask[i][t] bit b == (iou(sorted i, sorted t*32+b) > 0.5 && j>i)
// ---------------------------------------------------------------------------
__global__ __launch_bounds__(64) void nms_mask(const uint32_t* __restrict__ order_g,
                                               const float* __restrict__ obox_g,
                                               uint32_t* __restrict__ mask_g)
{
    const int i = blockIdx.x, t = threadIdx.x;
    uint32_t oi = order_g[i];
    float x1i = obox_g[oi*4+0], y1i = obox_g[oi*4+1];
    float x2i = obox_g[oi*4+2], y2i = obox_g[oi*4+3];
    float ai = (x2i - x1i) * (y2i - y1i);
    uint32_t msk = 0u;
    for (int b = 0; b < 32; ++b) {
        int j = t*32 + b;
        if (j > i) {
            uint32_t oj = order_g[j];
            float x1 = obox_g[oj*4+0], y1 = obox_g[oj*4+1];
            float x2 = obox_g[oj*4+2], y2 = obox_g[oj*4+3];
            float aj = (x2 - x1) * (y2 - y1);
            float xx1 = fmaxf(x1i, x1), yy1 = fmaxf(y1i, y1);
            float xx2 = fminf(x2i, x2), yy2 = fminf(y2i, y2);
            float inter = fmaxf(xx2 - xx1, 0.f) * fmaxf(yy2 - yy1, 0.f);
            float iou = inter / (ai + aj - inter + 1e-12f);
            if (iou > 0.5f) msk |= (1u << b);
        }
    }
    mask_g[i*64 + t] = msk;
}

// ---------------------------------------------------------------------------
// K5: greedy scan (single wave, lockstep) + write final output
// ---------------------------------------------------------------------------
__global__ __launch_bounds__(32) void nms_scan(const uint32_t* __restrict__ order_g,
                                               const uint32_t* __restrict__ mask_g,
                                               const float* __restrict__ score_g,
                                               const float* __restrict__ obox_g,
                                               float* __restrict__ out)
{
    volatile __shared__ uint32_t keep[64];
    const int t = threadIdx.x;
    for (int w = t; w < 64; w += 32) {
        uint32_t kw = 0u;
        for (int b = 0; b < 32; ++b) {
            uint32_t oj = order_g[w*32 + b];
            if (score_g[oj] >= 0.5f) kw |= (1u << b);
        }
        keep[w] = kw;
    }
    for (int i = 0; i < NBOX; ++i) {
        uint32_t kb = (keep[i >> 5] >> (i & 31)) & 1u;
        if (kb) {
            keep[t]      &= ~mask_g[i*64 + t];
            keep[t + 32] &= ~mask_g[i*64 + t + 32];
        }
    }
    for (int i = t; i < NBOX; i += 32) {
        uint32_t kb = (keep[i >> 5] >> (i & 31)) & 1u;
        uint32_t orig = order_g[i];
        for (int q = 0; q < 4; ++q)
            out[orig*4 + q] = kb ? obox_g[orig*4 + q] : 0.f;
    }
}

// ---------------------------------------------------------------------------
extern "C" void kernel_launch(void* const* d_in, const int* in_sizes, int n_in,
                              void* d_out, int out_size, void* d_ws, size_t ws_size,
                              hipStream_t stream)
{
    (void)in_sizes; (void)n_in; (void)out_size; (void)ws_size;
    const float* bboxes = (const float*)d_in[0];
    const float* image  = (const float*)d_in[1];
    const float* c1w = (const float*)d_in[2];
    const float* c1b = (const float*)d_in[3];
    const float* p1  = (const float*)d_in[4];
    const float* c2w = (const float*)d_in[5];
    const float* c2b = (const float*)d_in[6];
    const float* p2  = (const float*)d_in[7];
    const float* c3w = (const float*)d_in[8];
    const float* c3b = (const float*)d_in[9];
    const float* p3  = (const float*)d_in[10];
    const float* c4w = (const float*)d_in[11];
    const float* c4b = (const float*)d_in[12];
    const float* p4  = (const float*)d_in[13];
    const float* f5w = (const float*)d_in[14];
    const float* f5b = (const float*)d_in[15];
    const float* p5  = (const float*)d_in[16];
    const float* f6aw = (const float*)d_in[17];
    const float* f6ab = (const float*)d_in[18];
    const float* f6bw = (const float*)d_in[19];
    const float* f6bb = (const float*)d_in[20];
    float* out = (float*)d_out;

    char* w = (char*)d_ws;
    size_t off = 0;
    auto alloc = [&](size_t bytes) { void* p = w + off; off += (bytes + 255) & ~(size_t)255; return p; };
    _Float16* Bw2 = (_Float16*)alloc(18432ull * 2);     // [64][288]
    _Float16* Bw3 = (_Float16*)alloc(36864ull * 2);     // [64][576]
    _Float16* Bw4 = (_Float16*)alloc(32768ull * 2);     // [128][256]
    _Float16* Bw5 = (_Float16*)alloc(294912ull * 2);    // [256][1152]
    _Float16* h4g = (_Float16*)alloc((size_t)NBOX * 1152 * 2);
    float*    score_g = (float*)alloc(NBOX * 4);
    float*    obox_g  = (float*)alloc(NBOX * 16);
    uint32_t* order_g = (uint32_t*)alloc(NBOX * 4);
    uint32_t* mask_g  = (uint32_t*)alloc((size_t)NBOX * 64 * 4);

    // K0: weight prep
    wprep_transpose<<<(18432 + 255)/256, 256, 0, stream>>>(c2w, Bw2, 288, 64);
    wprep_transpose<<<(36864 + 255)/256, 256, 0, stream>>>(c3w, Bw3, 576, 64);
    wprep_transpose<<<(32768 + 255)/256, 256, 0, stream>>>(c4w, Bw4, 256, 128);
    wprep_direct   <<<(294912 + 255)/256, 256, 0, stream>>>(f5w, Bw5, 294912);

    // K1: per-crop CNN
    onet_cnn<<<NBOX, 256, 0, stream>>>(bboxes, image, c1w, c1b, p1,
                                       c2b, p2, c3b, p3, c4b, p4,
                                       Bw2, Bw3, Bw4, h4g);
    // K2: fc5/fc6
    onet_fc<<<NBOX/16, 256, 0, stream>>>(h4g, Bw5, f5b, p5,
                                         f6aw, f6ab, f6bw, f6bb,
                                         bboxes, score_g, obox_g);
    // K3-K5: NMS
    nms_sort<<<1, 1024, 0, stream>>>(score_g, order_g);
    nms_mask<<<NBOX, 64, 0, stream>>>(order_g, obox_g, mask_g);
    nms_scan<<<1, 32, 0, stream>>>(order_g, mask_g, score_g, obox_g, out);
}
